// RNN_aided_HMVAE_toeplitz_34703335752121
// MI455X (gfx1250) — matrix operand order repaired and checked by reading, not compiled, not run on hardware
//
#include <hip/hip_runtime.h>
#include <hip/hip_bf16.h>

// ---------------------------------------------------------------------------
// CDNA5 (gfx1250, wave32) implementation.
// All dense layers go through v_wmma_f32_16x16x32_bf16 (f32 accumulate).
// Staging paths are branch-free: clamped addresses + cndmask selects, so the
// global->LDS fill vectorizes instead of exec-mask branching per element.
// Toeplitz outputs (2 x 268MB) are produced by a pure streaming store kernel.
// ---------------------------------------------------------------------------

typedef __attribute__((ext_vector_type(16))) __bf16       v16bf;
typedef __attribute__((ext_vector_type(8)))  float        v8f;
typedef __attribute__((ext_vector_type(4)))  unsigned int uint4v;

union FragBF { uint4v q[2]; v16bf v; };

__device__ __forceinline__ unsigned short f2bf(float x) {
  unsigned int u = __builtin_bit_cast(unsigned int, x);
  unsigned int r = u + 0x7FFFu + ((u >> 16) & 1u);   // round-to-nearest-even
  return (unsigned short)(r >> 16);
}
__device__ __forceinline__ float sigmoidf(float x) { return 1.0f / (1.0f + __expf(-x)); }

#define BATCH 2048
#define NU    16

// ---------------------------------------------------------------------------
// GEMM: Y[row*ldr + col*ldc] = act(A(MxK) @ W(KxN) + bias[col])
// Block tile 128x64, 8 waves (4x2), each wave 32x32 = 2x2 WMMA 16x16x32 bf16.
// act: 0 = none, 1 = relu.  M is always a multiple of 128 here (M = 2048).
// ---------------------------------------------------------------------------
__global__ __launch_bounds__(256)
void gemm_bias_act(const float* __restrict__ A, const float* __restrict__ W,
                   const float* __restrict__ bias, float* __restrict__ Y,
                   int M, int N, int K, int ldr, int ldc, int act)
{
  __shared__ __align__(16) unsigned short As[128 * 40];  // [m][k], pad 40
  __shared__ __align__(16) unsigned short Bs[64 * 40];   // [n][k], pad 40

  const int tid  = threadIdx.x;
  const int lane = tid & 31;
  const int wave = tid >> 5;
  const int wr   = wave & 3;    // wave row: 4 along M (32 rows each)
  const int wc   = wave >> 2;   // wave col: 2 along N (32 cols each)
  const int m0   = blockIdx.x * 128;
  const int n0   = blockIdx.y * 64;
  const int half = lane >> 4;
  const int l16  = lane & 15;

  v8f acc[2][2];
#pragma unroll
  for (int i = 0; i < 2; ++i)
#pragma unroll
    for (int j = 0; j < 2; ++j)
#pragma unroll
      for (int r = 0; r < 8; ++r) acc[i][j][r] = 0.0f;

  const int arow = tid >> 1;          // 0..127
  const int aks  = (tid & 1) * 16;    // 0 / 16
  const int bcol = tid >> 2;          // 0..63
  const int bks  = (tid & 3) * 8;     // 0/8/16/24

  // Column clamp: cols >= N stage garbage but are never stored in the epilogue.
  const int nc = (n0 + bcol < N) ? (n0 + bcol) : (N - 1);
  const float* __restrict__ Wn = W + nc;
  const float* __restrict__ Arow = A + (long long)(m0 + arow) * K;

  unsigned short* const dstA = &As[arow * 40 + aks];
  unsigned short* const dstB = &Bs[bcol * 40 + bks];

  for (int kk = 0; kk < K; kk += 32) {
    if (kk + 32 <= K) {
      // ---- fast path: fully in-range, no guards (vectorizable loads) ----
      const float* __restrict__ Ap = Arow + kk + aks;
#pragma unroll
      for (int j = 0; j < 16; ++j) dstA[j] = f2bf(Ap[j]);
#pragma unroll
      for (int j = 0; j < 8; ++j)
        dstB[j] = f2bf(Wn[(long long)(kk + bks + j) * N]);
    } else {
      // ---- K tail: clamped address + cndmask select (branch-free) ----
      const int rem = K - kk;                       // 1..31
      const float* __restrict__ Ap = Arow + kk;
#pragma unroll
      for (int j = 0; j < 16; ++j) {
        int k  = aks + j;
        int kc = (k < rem) ? k : (rem - 1);         // always a valid load
        float v = Ap[kc];
        dstA[j] = f2bf((k < rem) ? v : 0.0f);
      }
#pragma unroll
      for (int j = 0; j < 8; ++j) {
        int k  = bks + j;
        int kc = (k < rem) ? k : (rem - 1);
        float v = Wn[(long long)(kk + kc) * N];
        dstB[j] = f2bf((k < rem) ? v : 0.0f);
      }
    }
    __syncthreads();

    // --- build fragments per ISA VGPR layouts ---
    FragBF af[2], bf[2];
#pragma unroll
    for (int mi = 0; mi < 2; ++mi) {
      const unsigned short* src = &As[(wr * 32 + mi * 16 + l16) * 40];
      af[mi].q[0] = *(const uint4v*)(src + 8 * half);        // K = 8h..8h+7
      af[mi].q[1] = *(const uint4v*)(src + 16 + 8 * half);   // K = 16+8h..
    }
#pragma unroll
    for (int ni = 0; ni < 2; ++ni) {
      const unsigned short* src = &Bs[(wc * 32 + ni * 16 + l16) * 40];
      bf[ni].q[0] = *(const uint4v*)(src + 8 * half);        // K = 8h..8h+7
      bf[ni].q[1] = *(const uint4v*)(src + 16 + 8 * half);   // K = 16+8h..
    }
#pragma unroll
    for (int mi = 0; mi < 2; ++mi)
#pragma unroll
      for (int ni = 0; ni < 2; ++ni)
        acc[mi][ni] = __builtin_amdgcn_wmma_f32_16x16x32_bf16(
            false, af[mi].v, false, bf[ni].v, (short)0, acc[mi][ni], false, false);
    __syncthreads();
  }

  // --- epilogue: bias + activation + strided store ---
#pragma unroll
  for (int ni = 0; ni < 2; ++ni) {
    const int col = n0 + wc * 32 + ni * 16 + l16;
    if (col >= N) continue;
    const float bv = bias[col];
#pragma unroll
    for (int mi = 0; mi < 2; ++mi) {
#pragma unroll
      for (int r = 0; r < 8; ++r) {
        const int row = m0 + wr * 32 + mi * 16 + r + 8 * half;
        float y = acc[mi][ni][r] + bv;
        if (act == 1) y = fmaxf(y, 0.0f);
        Y[(long long)row * ldr + (long long)col * ldc] = y;
      }
    }
  }
}

// ---------------------------------------------------------------------------
// Elementwise / reduction helpers
// ---------------------------------------------------------------------------
__global__ void fill_kernel(float* __restrict__ p, float v, int n) {
  int i = blockIdx.x * 256 + threadIdx.x;
  if (i < n) p[i] = v;
}

// zin = concat(x_u (B,64), z_prev (B,32)) -> (B,96)
__global__ void build_zin(const float* __restrict__ x, const float* __restrict__ zprev,
                          float* __restrict__ zin, int u) {
  int i = blockIdx.x * 256 + threadIdx.x;           // over B*96
  if (i >= BATCH * 96) return;
  int f = i % 96, b = i / 96;
  float v = (f < 64) ? x[((long long)b * 64 + f) * NU + u]   // x (B,2,32,U) flat
                     : zprev[b * 32 + (f - 64)];
  zin[i] = v;
}

// encoder gate + reparameterization (h == 1)
__global__ void enc_gate(const float* __restrict__ fb, const float* __restrict__ cb,
                         const float* __restrict__ kb, const float* __restrict__ t3,
                         const float* __restrict__ eps, float* __restrict__ z_u,
                         float* __restrict__ outZ, float* __restrict__ outMu,
                         float* __restrict__ outLv, int u) {
  int i = blockIdx.x * 256 + threadIdx.x;           // over B*32
  if (i >= BATCH * 32) return;
  int l = i & 31, b = i >> 5;
  float ns = sigmoidf(fb[i]) + sigmoidf(cb[i]) * tanhf(kb[i]);
  float g  = tanhf(ns);
  float mu = g * t3[b * 64 + l];
  float lv = g * t3[b * 64 + 32 + l];
  float z  = mu + eps[(long long)i * NU + u] * __expf(0.5f * lv);
  z_u[i] = z;
  long long o = (long long)i * NU + u;              // (B,L,U) layout
  outZ[o] = z; outMu[o] = mu; outLv[o] = lv;
}

// prior gate (h == 0, sigmoid(f) term dead)
__global__ void pri_gate(const float* __restrict__ cb, const float* __restrict__ kb,
                         const float* __restrict__ t3, float* __restrict__ outMu,
                         float* __restrict__ outLp, int u) {
  int i = blockIdx.x * 256 + threadIdx.x;
  if (i >= BATCH * 32) return;
  int l = i & 31, b = i >> 5;
  float g = tanhf(sigmoidf(cb[i]) * tanhf(kb[i]));
  long long o = (long long)i * NU + u;
  outMu[o] = g * t3[b * 64 + l];
  outLp[o] = g * t3[b * 64 + 32 + l];
}

// BatchNorm stats: one block per feature, reduce over B=2048 (biased var)
__global__ void bn_stats(const float* __restrict__ X, int Nf,
                         float* __restrict__ mean, float* __restrict__ invstd) {
  int f = blockIdx.x, tid = threadIdx.x;
  float s = 0.0f, s2 = 0.0f;
  for (int b = tid; b < BATCH; b += 256) {
    float v = X[(long long)b * Nf + f];
    s += v; s2 += v * v;
  }
  __shared__ float sh[256], sh2[256];
  sh[tid] = s; sh2[tid] = s2; __syncthreads();
  for (int o = 128; o > 0; o >>= 1) {
    if (tid < o) { sh[tid] += sh[tid + o]; sh2[tid] += sh2[tid + o]; }
    __syncthreads();
  }
  if (tid == 0) {
    float m = sh[0] * (1.0f / BATCH);
    float v = sh2[0] * (1.0f / BATCH) - m * m;
    mean[f] = m; invstd[f] = rsqrtf(v + 1e-5f);
  }
}

__global__ void bn_apply(float* __restrict__ X, int Nf, const float* __restrict__ mean,
                         const float* __restrict__ invstd, const float* __restrict__ g,
                         const float* __restrict__ e, int total) {
  int i = blockIdx.x * 256 + threadIdx.x;
  if (i >= total) return;
  int f = i % Nf;
  X[i] = (X[i] - mean[f]) * invstd[f] * g[f] + e[f];
}

// decoder head: Alpha (B,32) complex from a0raw (B,1), arraw (B,62)
__global__ void dec_head(const float* __restrict__ a0raw, const float* __restrict__ arraw,
                         float* __restrict__ alpha_u) {
  int b = blockIdx.x * 256 + threadIdx.x;
  if (b >= BATCH) return;
  float a0 = fminf(__expf(a0raw[b]), 500.0f);
  float2* Au = (float2*)(alpha_u + (long long)b * 64);
  Au[0] = make_float2(a0, 0.0f);
  const float* ar = arraw + (long long)b * 62;
#pragma unroll 4
  for (int j = 0; j < 31; ++j)
    Au[1 + j] = make_float2(0.022f * a0 * tanhf(ar[j]),
                            0.022f * a0 * tanhf(ar[31 + j]));
}

// Toeplitz writer: B[i][j]=Alpha[i-j] (j<=i); C[i][j]=conj(Alpha[32-(i-j)]) (j<i)
// Pure streaming store: 537 MB total -> HBM-bound floor (~23us at 23.3 TB/s).
__global__ void toeplitz_out(const float* __restrict__ alpha,
                             float* __restrict__ Bout, float* __restrict__ Cout) {
  long long idx = (long long)blockIdx.x * 256 + threadIdx.x;  // B*U*32*32
  if (idx >= (long long)BATCH * NU * 1024) return;
  int j = (int)(idx & 31);
  int i = (int)((idx >> 5) & 31);
  long long bu = idx >> 10;                                   // u*2048 + b
  const float2* Au = (const float2*)(alpha + bu * 64);
  int d = i - j;
  float2 bm = make_float2(0.0f, 0.0f), cm = make_float2(0.0f, 0.0f);
  if (d > 0) {
    bm = Au[d];
    float2 c = Au[32 - d];
    cm = make_float2(c.x, -c.y);
  } else if (d == 0) {
    bm = Au[0];
  }
  ((float2*)Bout)[idx] = bm;
  ((float2*)Cout)[idx] = cm;
}

// ---------------------------------------------------------------------------
// Host orchestration
// ---------------------------------------------------------------------------
extern "C" void kernel_launch(void* const* d_in, const int* in_sizes, int n_in,
                              void* d_out, int out_size, void* d_ws, size_t ws_size,
                              hipStream_t stream) {
  const float* x   = (const float*)d_in[0];
  const float* eps = (const float*)d_in[1];
  // encoder params (per-unit leading dim U=16)
  const float *eWf = (const float*)d_in[2],  *ebf = (const float*)d_in[3];
  const float *eWc = (const float*)d_in[4],  *ebc = (const float*)d_in[5];
  const float *eWk = (const float*)d_in[6],  *ebk = (const float*)d_in[7];
  const float *eW1 = (const float*)d_in[8],  *eb1 = (const float*)d_in[9];
  const float *eW2 = (const float*)d_in[10], *eb2 = (const float*)d_in[11];
  const float *eW3 = (const float*)d_in[12], *eb3 = (const float*)d_in[13];
  // prior params
  const float *pWc = (const float*)d_in[16], *pbc = (const float*)d_in[17];
  const float *pWk = (const float*)d_in[18], *pbk = (const float*)d_in[19];
  const float *pW1 = (const float*)d_in[20], *pb1 = (const float*)d_in[21];
  const float *pW2 = (const float*)d_in[22], *pb2 = (const float*)d_in[23];
  const float *pW3 = (const float*)d_in[24], *pb3 = (const float*)d_in[25];
  // decoder params
  const float *A1 = (const float*)d_in[26], *a1 = (const float*)d_in[27];
  const float *g1 = (const float*)d_in[28], *e1 = (const float*)d_in[29];
  const float *A2 = (const float*)d_in[30], *a2 = (const float*)d_in[31];
  const float *g2 = (const float*)d_in[32], *e2 = (const float*)d_in[33];
  const float *M1 = (const float*)d_in[34], *m1 = (const float*)d_in[35];
  const float *gm = (const float*)d_in[36], *em = (const float*)d_in[37];
  const float *M2 = (const float*)d_in[38], *m2 = (const float*)d_in[39];
  const float *P1 = (const float*)d_in[40], *p1 = (const float*)d_in[41];
  const float *gp = (const float*)d_in[42], *ep = (const float*)d_in[43];
  const float *P2 = (const float*)d_in[44], *p2 = (const float*)d_in[45];
  const float *Q1 = (const float*)d_in[46], *q1 = (const float*)d_in[47];
  const float *gq = (const float*)d_in[48], *eq = (const float*)d_in[49];
  const float *Q2 = (const float*)d_in[50], *q2 = (const float*)d_in[51];

  float* out = (float*)d_out;
  // output base offsets (floats; complex64 = 2 contiguous floats)
  const long long base_B   = 2097152LL;
  const long long base_C   = base_B + 67108864LL;
  const long long base_z   = base_C + 67108864LL;
  const long long base_mui = base_z   + 1048576LL;
  const long long base_lv  = base_mui + 1048576LL;
  const long long base_mup = base_lv  + 1048576LL;
  const long long base_lp  = base_mup + 1048576LL;

  // workspace carve-up (floats), ~27 MB total
  float* w = (float*)d_ws;
  float* ones  = w; w += 65536;
  float* zeros = w; w += 65536;
  float* zin   = w; w += 196608;          // (B,96)
  float* z_all = w; w += 1048576;         // (U,B,32)
  float* fbuf  = w; w += 65536;
  float* cbuf  = w; w += 65536;
  float* kbuf  = w; w += 65536;
  float* t1    = w; w += 819200;          // up to (B,400)
  float* t2    = w; w += 819200;
  float* t3    = w; w += 131072;          // (B,64)
  float* hbuf  = w; w += 819200;
  float* a0raw = w; w += 2048;
  float* arraw = w; w += 126976;          // (B,62)
  float* alpha = w; w += 2097152;         // (U,B,32) complex
  float* meanb = w; w += 512;
  float* istdb = w; w += 512;

  auto gemm = [&](const float* Ap, const float* Wp, const float* bp, float* Yp,
                  int M, int N, int K, int ldr, int ldc, int act) {
    dim3 grid((M + 127) / 128, (N + 63) / 64, 1);
    gemm_bias_act<<<grid, 256, 0, stream>>>(Ap, Wp, bp, Yp, M, N, K, ldr, ldc, act);
  };
  const int EB = (BATCH * 32 + 255) / 256;

  fill_kernel<<<(65536 + 255) / 256, 256, 0, stream>>>(ones, 1.0f, 65536);
  fill_kernel<<<(65536 + 255) / 256, 256, 0, stream>>>(zeros, 0.0f, 65536);

  // ---- encoder: sequential over units ----
  for (int u = 0; u < NU; ++u) {
    const float* zp = (u == 0) ? ones : (z_all + (long long)(u - 1) * 65536);
    build_zin<<<(BATCH * 96 + 255) / 256, 256, 0, stream>>>(x, zp, zin, u);
    gemm(zp, eWf + u * 1024, ebf + u * 32, fbuf, BATCH, 32, 32, 32, 1, 0);
    gemm(zp, eWc + u * 1024, ebc + u * 32, cbuf, BATCH, 32, 32, 32, 1, 0);
    gemm(zp, eWk + u * 1024, ebk + u * 32, kbuf, BATCH, 32, 32, 32, 1, 0);
    gemm(zin, eW1 + u * 28800, eb1 + u * 300, t1, BATCH, 300, 96, 300, 1, 1);
    gemm(t1,  eW2 + u * 60000, eb2 + u * 200, t2, BATCH, 200, 300, 200, 1, 1);
    gemm(t2,  eW3 + u * 12800, eb3 + u * 64,  t3, BATCH, 64, 200, 64, 1, 0);
    enc_gate<<<EB, 256, 0, stream>>>(fbuf, cbuf, kbuf, t3, eps,
                                     z_all + (long long)u * 65536,
                                     out + base_z, out + base_mui, out + base_lv, u);
  }

  // ---- prior: independent per unit (consumes z_{u-1}, zeros at u=0) ----
  for (int u = 0; u < NU; ++u) {
    const float* zp = (u == 0) ? zeros : (z_all + (long long)(u - 1) * 65536);
    gemm(zp, pWc + u * 1024, pbc + u * 32, cbuf, BATCH, 32, 32, 32, 1, 0);
    gemm(zp, pWk + u * 1024, pbk + u * 32, kbuf, BATCH, 32, 32, 32, 1, 0);
    gemm(zp, pW1 + u * 6400,  pb1 + u * 200, t1, BATCH, 200, 32, 200, 1, 1);
    gemm(t1, pW2 + u * 40000, pb2 + u * 200, t2, BATCH, 200, 200, 200, 1, 1);
    gemm(t2, pW3 + u * 12800, pb3 + u * 64,  t3, BATCH, 64, 200, 64, 1, 0);
    pri_gate<<<EB, 256, 0, stream>>>(cbuf, kbuf, t3, out + base_mup, out + base_lp, u);
  }

  // ---- decoder: independent per unit ----
  for (int u = 0; u < NU; ++u) {
    const float* zu = z_all + (long long)u * 65536;
    // out1 = BN(relu(z @ A1 + a1))  (B,200)
    gemm(zu, A1 + u * 6400, a1 + u * 200, t1, BATCH, 200, 32, 200, 1, 1);
    bn_stats<<<200, 256, 0, stream>>>(t1, 200, meanb, istdb);
    bn_apply<<<(BATCH * 200 + 255) / 256, 256, 0, stream>>>(t1, 200, meanb, istdb,
                                                            g1 + u * 200, e1 + u * 200, BATCH * 200);
    // out2 = BN(relu(out1 @ A2 + a2))  (B,400)
    gemm(t1, A2 + u * 80000, a2 + u * 400, t2, BATCH, 400, 200, 400, 1, 1);
    bn_stats<<<400, 256, 0, stream>>>(t2, 400, meanb, istdb);
    bn_apply<<<(BATCH * 400 + 255) / 256, 256, 0, stream>>>(t2, 400, meanb, istdb,
                                                            g2 + u * 400, e2 + u * 400, BATCH * 400);
    // mu branch -> writes directly into mu_out (B,2,32,U) via strided store
    gemm(t2, M1 + u * 160000, m1 + u * 400, hbuf, BATCH, 400, 400, 400, 1, 1);
    bn_stats<<<400, 256, 0, stream>>>(hbuf, 400, meanb, istdb);
    bn_apply<<<(BATCH * 400 + 255) / 256, 256, 0, stream>>>(hbuf, 400, meanb, istdb,
                                                            gm + u * 400, em + u * 400, BATCH * 400);
    gemm(hbuf, M2 + u * 25600, m2 + u * 64, out + u, BATCH, 64, 400, 1024, 16, 0);
    // a0 branch
    gemm(t2, P1 + u * 60000, p1 + u * 150, hbuf, BATCH, 150, 400, 150, 1, 1);
    bn_stats<<<150, 256, 0, stream>>>(hbuf, 150, meanb, istdb);
    bn_apply<<<(BATCH * 150 + 255) / 256, 256, 0, stream>>>(hbuf, 150, meanb, istdb,
                                                            gp + u * 150, ep + u * 150, BATCH * 150);
    gemm(hbuf, P2 + u * 150, p2 + u, a0raw, BATCH, 1, 150, 1, 1, 0);
    // ar branch
    gemm(t2, Q1 + u * 120000, q1 + u * 300, hbuf, BATCH, 300, 400, 300, 1, 1);
    bn_stats<<<300, 256, 0, stream>>>(hbuf, 300, meanb, istdb);
    bn_apply<<<(BATCH * 300 + 255) / 256, 256, 0, stream>>>(hbuf, 300, meanb, istdb,
                                                            gq + u * 300, eq + u * 300, BATCH * 300);
    gemm(hbuf, Q2 + u * 18600, q2 + u * 62, arraw, BATCH, 62, 300, 62, 1, 0);
    dec_head<<<(BATCH + 255) / 256, 256, 0, stream>>>(a0raw, arraw,
                                                      alpha + (long long)u * 131072);
  }

  // ---- Toeplitz materialization (bandwidth-bound output) ----
  const long long tot = (long long)BATCH * NU * 1024;
  toeplitz_out<<<(unsigned)((tot + 255) / 256), 256, 0, stream>>>(alpha, out + base_B, out + base_C);
}
